// ForgettingTransformerBlock_74775380623510
// MI455X (gfx1250) — matrix operand order repaired and checked by reading, not compiled, not run on hardware
//
#include <hip/hip_runtime.h>
#include <hip/hip_bf16.h>
#include <math.h>

#define B_  2
#define S_  2048
#define D_  1024
#define H_  16
#define DH_ 64
#define FF_ 4096
#define BS_ (B_*S_)

typedef __attribute__((ext_vector_type(16))) __bf16 v16bf;
typedef __attribute__((ext_vector_type(8)))  float  v8f;

union FragAB { v16bf v; unsigned u[8]; };
union FragC  { v8f   v; float    f[8]; };

__device__ __forceinline__ unsigned short f2bf(float x) {
  union { float f; unsigned u; } a; a.f = x;
  unsigned r = a.u + (0x7FFFu + ((a.u >> 16) & 1u));
  return (unsigned short)(r >> 16);
}
__device__ __forceinline__ float bf2f(unsigned short h) {
  union { unsigned u; float f; } a; a.u = ((unsigned)h) << 16;
  return a.f;
}
// 16-bit A-matrix 16x32 VGPR layout (ISA 7.12.2): k index for VGPR r, lane-half h
__device__ __forceinline__ int kmapA(int r, int half) {
  return ((r >> 2) << 4) + ((r & 3) << 1) + (half << 3);
}

// ---------- gfx1250 async global->LDS staging (ASYNCcnt path), with fallback ----------
#if defined(__AMDGCN__) && __has_builtin(__builtin_amdgcn_global_load_async_to_lds_b128)
#define HAS_ASYNC 1
#else
#define HAS_ASYNC 0
#endif

typedef int v4i_gcc __attribute__((vector_size(16)));
#define AS1_ __attribute__((address_space(1)))
#define AS3_ __attribute__((address_space(3)))

__device__ __forceinline__ void stage_b128(unsigned short* lds_dst,
                                           const unsigned short* gsrc) {
#if HAS_ASYNC
  // flat->AS1 is identity; flat->AS3 is low-32-bit truncation (ISA 10.2 aperture rules)
  __builtin_amdgcn_global_load_async_to_lds_b128(
      (AS1_ v4i_gcc*)(unsigned long long)gsrc,
      (AS3_ v4i_gcc*)(unsigned)(unsigned long long)lds_dst,
      0, 0);
#else
  *(uint4*)lds_dst = *(const uint4*)gsrc;
#endif
}
__device__ __forceinline__ void async_wait_all() {
#if HAS_ASYNC
  asm volatile("s_wait_asynccnt 0" ::: "memory");
#endif
}

// ---------------- elementwise f32 -> bf16 (weights) ----------------
__global__ void cvt_f32_bf16(const float* __restrict__ in,
                             unsigned short* __restrict__ out, int n) {
  int i = blockIdx.x * blockDim.x + threadIdx.x;
  if (i < n) out[i] = f2bf(in[i]);
}

// ---------------- LayerNorm (f32 in, bf16 out), one row per block ----------------
__global__ __launch_bounds__(256) void layernorm_bf16(
    const float* __restrict__ x, const float* __restrict__ g,
    const float* __restrict__ b, unsigned short* __restrict__ out) {
  __shared__ float red[256];
  const int row = blockIdx.x, tid = threadIdx.x;
  const float* xr = x + (size_t)row * D_;
  float s = 0.f;
  for (int i = tid; i < D_; i += 256) s += xr[i];
  red[tid] = s; __syncthreads();
  for (int o = 128; o > 0; o >>= 1) { if (tid < o) red[tid] += red[tid + o]; __syncthreads(); }
  float mu = red[0] / (float)D_;
  __syncthreads();
  float s2 = 0.f;
  for (int i = tid; i < D_; i += 256) { float d = xr[i] - mu; s2 += d * d; }
  red[tid] = s2; __syncthreads();
  for (int o = 128; o > 0; o >>= 1) { if (tid < o) red[tid] += red[tid + o]; __syncthreads(); }
  float rstd = rsqrtf(red[0] / (float)D_ + 1e-5f);
  for (int i = tid; i < D_; i += 256)
    out[(size_t)row * D_ + i] = f2bf((xr[i] - mu) * rstd * g[i] + b[i]);
}

// ---------------- generic bf16 WMMA GEMM: out = epi(A@W + bias) ----------------
// block tile 128x128, 8 waves (4M x 2N), wave tile 32x64 -> 8 WMMA per K-step.
// Double-buffered LDS: async A copy + transposed B staging for tile t+1 overlap
// with the WMMA work on tile t.
enum { EPI_BF16 = 0, EPI_GELU = 1, EPI_RES = 2 };

template <int EPI>
__global__ __launch_bounds__(256) void gemm_bf16_wmma(
    const unsigned short* __restrict__ A,   // M x K, row-major bf16
    const unsigned short* __restrict__ W,   // K x N, row-major bf16
    const float* __restrict__ bias,         // N
    const float* __restrict__ resid,        // M x N (EPI_RES)
    unsigned short* __restrict__ outb,      // M x N bf16 (EPI_BF16/GELU)
    float* __restrict__ outf,               // M x N f32  (EPI_RES)
    int M, int N, int K) {
  __shared__ __align__(16) unsigned short As[2][128 * 32];
  __shared__ __align__(16) unsigned short Bs[2][128 * 32];  // transposed [n][k]
  const int tid = threadIdx.x;
  const int lane = tid & 31;
  const int wave = __builtin_amdgcn_readfirstlane(tid >> 5);  // uniform wave id
  const int l16 = lane & 15, half = lane >> 4;
  const int wm = wave >> 1, wn = wave & 1;               // 4(M) x 2(N) wave grid
  const int m0 = blockIdx.x * 128, n0 = blockIdx.y * 128;
  (void)M;

  auto stageA = [&](int k0, int buf) {
#pragma unroll
    for (int ch = 0; ch < 2; ++ch) {
      int cid = tid + ch * 256;
      int row = cid >> 2, col = (cid & 3) << 3;
      stage_b128(&As[buf][row * 32 + col], A + (size_t)(m0 + row) * K + k0 + col);
    }
  };
  auto stageB = [&](int k0, int buf) {
#pragma unroll
    for (int ch = 0; ch < 2; ++ch) {
      int cid = tid + ch * 256;
      int kk = cid >> 4, nn = (cid & 15) << 3;
      union { uint4 q; unsigned short s[8]; } d;
      d.q = *(const uint4*)(W + (size_t)(k0 + kk) * N + n0 + nn);
#pragma unroll
      for (int e = 0; e < 8; ++e) Bs[buf][(nn + e) * 32 + kk] = d.s[e];
    }
  };

  FragC c[2][4];
#pragma unroll
  for (int i = 0; i < 2; ++i)
#pragma unroll
    for (int j = 0; j < 4; ++j)
#pragma unroll
      for (int r = 0; r < 8; ++r) c[i][j].f[r] = 0.f;

  stageA(0, 0);
  stageB(0, 0);
  for (int k0 = 0; k0 < K; k0 += 32) {
    const int buf = (k0 >> 5) & 1;
    async_wait_all();      // our async copies for this tile are done
    __syncthreads();       // everyone's copies (async + DS stores) are done
    if (k0 + 32 < K) {     // overlap next-tile staging with this tile's WMMAs
      stageA(k0 + 32, buf ^ 1);
      stageB(k0 + 32, buf ^ 1);
    }

    FragAB a[2], bfr[4];
#pragma unroll
    for (int i = 0; i < 2; ++i) {
      int m = wm * 32 + i * 16 + l16;
#pragma unroll
      for (int r = 0; r < 8; ++r)
        a[i].u[r] = *(const unsigned*)(&As[buf][m * 32 + kmapA(r, half)]);
    }
#pragma unroll
    for (int j = 0; j < 4; ++j) {
      int n = wn * 64 + j * 16 + l16;
#pragma unroll
      for (int r = 0; r < 8; ++r)
        bfr[j].u[r] = *(const unsigned*)(&Bs[buf][n * 32 + kmapA(r, half)]);
    }
#pragma unroll
    for (int i = 0; i < 2; ++i)
#pragma unroll
      for (int j = 0; j < 4; ++j)
        c[i][j].v = __builtin_amdgcn_wmma_f32_16x16x32_bf16(
            false, a[i].v, false, bfr[j].v, (short)0, c[i][j].v, false, false);
  }

#pragma unroll
  for (int i = 0; i < 2; ++i)
#pragma unroll
    for (int j = 0; j < 4; ++j)
#pragma unroll
      for (int r = 0; r < 8; ++r) {
        int row = m0 + wm * 32 + i * 16 + (half << 3) + r;
        int col = n0 + wn * 64 + j * 16 + l16;
        float v = c[i][j].f[r] + bias[col];
        if (EPI == EPI_GELU) {
          v = 0.5f * v * (1.f + erff(v * 0.70710678118f));
          outb[(size_t)row * N + col] = f2bf(v);
        } else if (EPI == EPI_RES) {
          outf[(size_t)row * N + col] = resid[(size_t)row * N + col] + v;
        } else {
          outb[(size_t)row * N + col] = f2bf(v);
        }
      }
}

// ---------------- forget gate logits: logf = log_sigmoid(h @ wf + bf) ----------------
__global__ __launch_bounds__(256) void forget_logits(
    const unsigned short* __restrict__ h, const float* __restrict__ wf,
    const float* __restrict__ bfv, float* __restrict__ lf) {
  int lane = threadIdx.x & 31, wave = threadIdx.x >> 5;
  int gw = blockIdx.x * 8 + wave;            // one wave per (row, head)
  int row = gw >> 4, head = gw & 15;
  const unsigned short* hr = h + (size_t)row * D_;
  float s = 0.f;
  for (int d = lane; d < D_; d += 32) s += bf2f(hr[d]) * wf[d * H_ + head];
  for (int o = 16; o > 0; o >>= 1) s += __shfl_xor(s, o, 32);
  if (lane == 0) {
    float x = s + bfv[head];
    lf[(size_t)row * H_ + head] = fminf(x, 0.f) - log1pf(__expf(-fabsf(x)));
  }
}

// ---------------- cumulative sum over S per (b, h), in place ----------------
__global__ __launch_bounds__(256) void cumsum_seq(float* __restrict__ c) {
  __shared__ float ts[256];
  int bh = blockIdx.x, b = bh / H_, hd = bh % H_;
  int tid = threadIdx.x;
  float* base = c + (size_t)b * S_ * H_ + hd;
  float loc[8];
  float acc = 0.f;
#pragma unroll
  for (int e = 0; e < 8; ++e) { acc += base[(size_t)(tid * 8 + e) * H_]; loc[e] = acc; }
  ts[tid] = acc; __syncthreads();
  if (tid == 0) {
    float a = 0.f;
    for (int i = 0; i < 256; ++i) { float t = ts[i]; ts[i] = a; a += t; }
  }
  __syncthreads();
  float pre = ts[tid];
#pragma unroll
  for (int e = 0; e < 8; ++e) base[(size_t)(tid * 8 + e) * H_] = pre + loc[e];
}

// ---------------- flash-style causal attention with forgetting decay ----------------
// block = 4 waves x 16 query rows. K staged via async copy; V staged TRANSPOSED
// (Vt[d][key]) so B-fragments are plain b32 DS reads. Double-buffered LDS:
// next key tile staged while current tile's WMMAs run. Wave-uniform (SGPR) key
// loop control; causal masking only on the 1-2 diagonal tiles per wave.
__global__ __launch_bounds__(128) void attn_forget(
    const unsigned short* __restrict__ q, const unsigned short* __restrict__ k,
    const unsigned short* __restrict__ v, const float* __restrict__ c,
    unsigned short* __restrict__ ctx) {
  __shared__ __align__(16) unsigned short Ks[2][32 * 64];  // [key][d]
  __shared__ __align__(16) unsigned short Vt[2][64 * 32];  // [d][key] (transposed)
  __shared__ __align__(16) unsigned short P[4][16 * 32];   // per-wave P bounce buffer
  const int tid = threadIdx.x, lane = tid & 31;
  const int wave = __builtin_amdgcn_readfirstlane(tid >> 5);  // uniform wave id
  const int l16 = lane & 15, half = lane >> 4;
  const int bh = blockIdx.y, b = bh >> 4, hd = bh & 15;
  const int q0 = blockIdx.x * 64 + wave * 16;             // uniform: 16 rows per wave
  const int jmax = blockIdx.x * 64 + 63;                  // block-uniform key bound
  const unsigned short* qb = q + (size_t)b * S_ * D_ + hd * DH_;
  const unsigned short* kb = k + (size_t)b * S_ * D_ + hd * DH_;
  const unsigned short* vb = v + (size_t)b * S_ * D_ + hd * DH_;
  const float* cb = c + (size_t)b * S_ * H_ + hd;
  unsigned short* Pw = P[wave];

  auto stageKV = [&](int j0, int buf) {
#pragma unroll
    for (int ch = 0; ch < 2; ++ch) {
      int elem = (tid + ch * 128) * 8;   // 0..2040, 8 bf16 per chunk
      int key = elem >> 6, d0 = elem & 63;
      stage_b128(&Ks[buf][elem], kb + (size_t)(j0 + key) * D_ + d0);
      union { uint4 q4; unsigned short s[8]; } dv;
      dv.q4 = *(const uint4*)(vb + (size_t)(j0 + key) * D_ + d0);
#pragma unroll
      for (int e = 0; e < 8; ++e) Vt[buf][(d0 + e) * 32 + key] = dv.s[e];
    }
  };

  FragAB qa[2];
#pragma unroll
  for (int t = 0; t < 2; ++t)
#pragma unroll
    for (int r = 0; r < 8; ++r)
      qa[t].u[r] = *(const unsigned*)(qb + (size_t)(q0 + l16) * D_ + t * 32 + kmapA(r, half));

  float ci[8], mrow[8], lrow[8];
  FragC acc[4];
#pragma unroll
  for (int r = 0; r < 8; ++r) {
    ci[r] = cb[(size_t)(q0 + (half << 3) + r) * H_];
    mrow[r] = -1e30f;
    lrow[r] = 0.f;
  }
#pragma unroll
  for (int t = 0; t < 4; ++t)
#pragma unroll
    for (int r = 0; r < 8; ++r) acc[t].f[r] = 0.f;

  const float scale = 0.125f;  // 1/sqrt(64)
  stageKV(0, 0);
  for (int j0 = 0; j0 <= jmax; j0 += 32) {
    const int buf = (j0 >> 5) & 1;
    async_wait_all();          // our async K copies done
    __syncthreads();           // everyone's K/V staging for this tile done
    if (j0 + 32 <= jmax) stageKV(j0 + 32, buf ^ 1);  // overlap with compute
    if (j0 > q0 + 15) continue;  // uniform scalar branch: rows all masked here

    // --- S = Q K^T for a 16x32 key tile (two 16x16 WMMA C tiles) ---
    FragC sc[2];
#pragma unroll
    for (int jt = 0; jt < 2; ++jt) {
      FragAB bk[2];
#pragma unroll
      for (int t = 0; t < 2; ++t)
#pragma unroll
        for (int r = 0; r < 8; ++r)
          bk[t].u[r] = *(const unsigned*)(&Ks[buf][(jt * 16 + l16) * 64 + t * 32 +
                                                   kmapA(r, half)]);
      v8f z = {};
      sc[jt].v = __builtin_amdgcn_wmma_f32_16x16x32_bf16(false, qa[0].v, false, bk[0].v,
                                                         (short)0, z, false, false);
      sc[jt].v = __builtin_amdgcn_wmma_f32_16x16x32_bf16(false, qa[1].v, false, bk[1].v,
                                                         (short)0, sc[jt].v, false, false);
    }
    float cj0 = cb[(size_t)(j0 + l16) * H_];
    float cj1 = cb[(size_t)(j0 + 16 + l16) * H_];

    // --- scale + decay bias (+ causal mask on diagonal tiles only) ---
    float tmax[8];
    if (j0 + 31 <= q0) {
      // fast path: tile fully below the diagonal, no masking needed
#pragma unroll
      for (int r = 0; r < 8; ++r) {
        float s0 = sc[0].f[r] * scale + ci[r] - cj0;
        float s1 = sc[1].f[r] * scale + ci[r] - cj1;
        sc[0].f[r] = s0; sc[1].f[r] = s1;
        float t = fmaxf(s0, s1);
        for (int o = 1; o < 16; o <<= 1) t = fmaxf(t, __shfl_xor(t, o, 32));
        tmax[r] = t;
      }
    } else {
      // diagonal tile: apply causal mask
#pragma unroll
      for (int r = 0; r < 8; ++r) {
        int i = q0 + (half << 3) + r;
        float s0 = sc[0].f[r] * scale + ci[r] - cj0;
        float s1 = sc[1].f[r] * scale + ci[r] - cj1;
        if (j0 + l16 > i) s0 = -1e30f;
        if (j0 + 16 + l16 > i) s1 = -1e30f;
        sc[0].f[r] = s0; sc[1].f[r] = s1;
        float t = fmaxf(s0, s1);
        for (int o = 1; o < 16; o <<= 1) t = fmaxf(t, __shfl_xor(t, o, 32));
        tmax[r] = t;
      }
    }
    // --- online softmax update ---
    float alpha[8];
#pragma unroll
    for (int r = 0; r < 8; ++r) {
      float mn = fmaxf(mrow[r], tmax[r]);
      alpha[r] = __expf(mrow[r] - mn);
      mrow[r] = mn;
      float p0 = __expf(sc[0].f[r] - mn);
      float p1 = __expf(sc[1].f[r] - mn);
      sc[0].f[r] = p0; sc[1].f[r] = p1;
      float rs = p0 + p1;
      for (int o = 1; o < 16; o <<= 1) rs += __shfl_xor(rs, o, 32);
      lrow[r] = lrow[r] * alpha[r] + rs;
    }
#pragma unroll
    for (int t = 0; t < 4; ++t)
#pragma unroll
      for (int r = 0; r < 8; ++r) acc[t].f[r] *= alpha[r];

    // --- P: C layout -> A layout via per-wave LDS (same-wave DS is in order) ---
#pragma unroll
    for (int jt = 0; jt < 2; ++jt)
#pragma unroll
      for (int r = 0; r < 8; ++r)
        Pw[((half << 3) + r) * 32 + jt * 16 + l16] = f2bf(sc[jt].f[r]);
    FragAB pa;
#pragma unroll
    for (int r = 0; r < 8; ++r)
      pa.u[r] = *(const unsigned*)(Pw + l16 * 32 + kmapA(r, half));

    // --- ctx += P @ V (V fragments: aligned b32 reads from transposed Vt) ---
#pragma unroll
    for (int dt = 0; dt < 4; ++dt) {
      FragAB bv;
#pragma unroll
      for (int r = 0; r < 8; ++r)
        bv.u[r] = *(const unsigned*)(&Vt[buf][(dt * 16 + l16) * 32 +
                                              kmapA(r, half)]);
      acc[dt].v = __builtin_amdgcn_wmma_f32_16x16x32_bf16(false, pa.v, false, bv.v,
                                                          (short)0, acc[dt].v, false, false);
    }
  }
  // --- normalize and store bf16 ctx ---
#pragma unroll
  for (int dt = 0; dt < 4; ++dt)
#pragma unroll
    for (int r = 0; r < 8; ++r) {
      int row = q0 + (half << 3) + r;
      int col = hd * DH_ + dt * 16 + l16;
      ctx[((size_t)b * S_ + row) * D_ + col] = f2bf(acc[dt].f[r] / lrow[r]);
    }
}

// ---------------- host-side launch ----------------
extern "C" void kernel_launch(void* const* d_in, const int* in_sizes, int n_in,
                              void* d_out, int out_size, void* d_ws, size_t ws_size,
                              hipStream_t stream) {
  (void)in_sizes; (void)n_in; (void)out_size; (void)ws_size;
  const float* x    = (const float*)d_in[0];
  const float* ln1g = (const float*)d_in[1];
  const float* ln1b = (const float*)d_in[2];
  const float* wq = (const float*)d_in[3];  const float* bq = (const float*)d_in[4];
  const float* wk = (const float*)d_in[5];  const float* bk = (const float*)d_in[6];
  const float* wv = (const float*)d_in[7];  const float* bv = (const float*)d_in[8];
  const float* wo = (const float*)d_in[9];  const float* bo = (const float*)d_in[10];
  const float* wf = (const float*)d_in[11]; const float* bfv = (const float*)d_in[12];
  const float* ln2g = (const float*)d_in[13];
  const float* ln2b = (const float*)d_in[14];
  const float* w1 = (const float*)d_in[15]; const float* b1 = (const float*)d_in[16];
  const float* w2 = (const float*)d_in[17]; const float* b2 = (const float*)d_in[18];
  float* out = (float*)d_out;

  char* ws = (char*)d_ws;
  size_t off = 0;
  auto alloc = [&](size_t bytes) -> void* {
    void* p = ws + off;
    off += (bytes + 255) & ~(size_t)255;
    return p;
  };
  unsigned short* h1   = (unsigned short*)alloc((size_t)BS_ * D_ * 2);
  unsigned short* qbuf = (unsigned short*)alloc((size_t)BS_ * D_ * 2);
  unsigned short* kbuf = (unsigned short*)alloc((size_t)BS_ * D_ * 2);
  unsigned short* vbuf = (unsigned short*)alloc((size_t)BS_ * D_ * 2);
  unsigned short* ctxb = (unsigned short*)alloc((size_t)BS_ * D_ * 2);
  unsigned short* h2   = (unsigned short*)alloc((size_t)BS_ * D_ * 2);
  unsigned short* ff1  = (unsigned short*)alloc((size_t)BS_ * FF_ * 2);
  float* x1   = (float*)alloc((size_t)BS_ * D_ * 4);
  float* cbuf = (float*)alloc((size_t)BS_ * H_ * 4);
  unsigned short* wqb = (unsigned short*)alloc((size_t)D_ * D_ * 2);
  unsigned short* wkb = (unsigned short*)alloc((size_t)D_ * D_ * 2);
  unsigned short* wvb = (unsigned short*)alloc((size_t)D_ * D_ * 2);
  unsigned short* wob = (unsigned short*)alloc((size_t)D_ * D_ * 2);
  unsigned short* w1b = (unsigned short*)alloc((size_t)D_ * FF_ * 2);
  unsigned short* w2b = (unsigned short*)alloc((size_t)FF_ * D_ * 2);

  const int nDD = D_ * D_, nDF = D_ * FF_;
  cvt_f32_bf16<<<(nDD + 255) / 256, 256, 0, stream>>>(wq, wqb, nDD);
  cvt_f32_bf16<<<(nDD + 255) / 256, 256, 0, stream>>>(wk, wkb, nDD);
  cvt_f32_bf16<<<(nDD + 255) / 256, 256, 0, stream>>>(wv, wvb, nDD);
  cvt_f32_bf16<<<(nDD + 255) / 256, 256, 0, stream>>>(wo, wob, nDD);
  cvt_f32_bf16<<<(nDF + 255) / 256, 256, 0, stream>>>(w1, w1b, nDF);
  cvt_f32_bf16<<<(nDF + 255) / 256, 256, 0, stream>>>(w2, w2b, nDF);

  // LN1
  layernorm_bf16<<<BS_, 256, 0, stream>>>(x, ln1g, ln1b, h1);

  // Q, K, V projections
  dim3 gD(BS_ / 128, D_ / 128);
  gemm_bf16_wmma<EPI_BF16><<<gD, 256, 0, stream>>>(h1, wqb, bq, nullptr, qbuf, nullptr, BS_, D_, D_);
  gemm_bf16_wmma<EPI_BF16><<<gD, 256, 0, stream>>>(h1, wkb, bk, nullptr, kbuf, nullptr, BS_, D_, D_);
  gemm_bf16_wmma<EPI_BF16><<<gD, 256, 0, stream>>>(h1, wvb, bv, nullptr, vbuf, nullptr, BS_, D_, D_);

  // forgetting gate: log-sigmoid logits then cumulative decay
  forget_logits<<<BS_ * H_ / 8, 256, 0, stream>>>(h1, wf, bfv, cbuf);
  cumsum_seq<<<B_ * H_, 256, 0, stream>>>(cbuf);

  // attention
  dim3 gA(S_ / 64, B_ * H_);
  attn_forget<<<gA, 128, 0, stream>>>(qbuf, kbuf, vbuf, cbuf, ctxb);

  // O projection + residual (f32)
  gemm_bf16_wmma<EPI_RES><<<gD, 256, 0, stream>>>(ctxb, wob, bo, x, nullptr, x1, BS_, D_, D_);

  // LN2
  layernorm_bf16<<<BS_, 256, 0, stream>>>(x1, ln2g, ln2b, h2);

  // MLP
  dim3 gF(BS_ / 128, FF_ / 128);
  gemm_bf16_wmma<EPI_GELU><<<gF, 256, 0, stream>>>(h2, w1b, b1, nullptr, ff1, nullptr, BS_, FF_, D_);
  gemm_bf16_wmma<EPI_RES><<<gD, 256, 0, stream>>>(ff1, w2b, b2, x1, nullptr, out, BS_, D_, FF_);
}
